// HGT_10153302688437
// MI455X (gfx1250) — matrix operand order repaired and checked by reading, not compiled, not run on hardware
//
#include <hip/hip_runtime.h>
#include <hip/hip_bf16.h>
#include <math.h>

// ---------------------------------------------------------------------------
// HGT forward for MI455X (gfx1250, wave32, WMMA).
// Dense projections: bf16-in / f32-accum WMMA GEMMs (bandwidth bound at
// 23.3 TB/s). Weight fragments staged once per block in LDS (8 waves share
// 32 KB). Per-lane A row (256 B) loaded in a single clause up front, converted
// with native v_cvt_pk_bf16_f32, then 32 v_wmma run purely off LDS+registers.
// Per-edge relA/relM einsums folded into effective projection weights
// (kW_eff = kW @ blockdiag(relA)), turning edge einsums into node GEMMs.
// Softmax-attention uses float atomics (max / add) over destination nodes.
// ---------------------------------------------------------------------------

#define HH 4
#define DD 32
#define CC 128
#define NEDGE 3
#define LL 2
#define TT 2

typedef __bf16 bf16_t;
typedef bf16_t v16bf __attribute__((ext_vector_type(16)));
typedef float  v8f   __attribute__((ext_vector_type(8)));

union FragBF { v16bf v; unsigned short u[16]; uint4 q[2]; };

__device__ __forceinline__ unsigned short f2bf_bits(float f) {
  unsigned int u = __float_as_uint(f);
  unsigned int r = u + 0x7FFFu + ((u >> 16) & 1u);  // RNE (pack kernel only)
  return (unsigned short)(r >> 16);
}

__device__ __forceinline__ float gelu_tanh(float x) {
  float x3 = x * x * x;
  float t = tanhf(0.7978845608028654f * (x + 0.044715f * x3));
  return 0.5f * x * (1.0f + t);
}

__device__ __forceinline__ void atomicMaxF(float* addr, float val) {
  // ordered-int trick; m buffer initialized to -inf
  if (val >= 0.0f) atomicMax((int*)addr, __float_as_int(val));
  else             atomicMin((unsigned int*)addr, (unsigned int)__float_as_int(val));
}

// ---------------------------------------------------------------------------
// Workhorse: Y[M,128] = epilogue( act_pre(X)[M,128] @ W(128x128,bf16 frags) + b )
// One wave = one 16-row x 128-col tile (8 x v8f accumulators, 32 v_wmma).
// Block = 256 threads = 8 waves = 128 rows. B fragments served from LDS.
// ---------------------------------------------------------------------------
template<int PRE_GELU, int POST_RELU, int SKIP_MIX>
__global__ void __launch_bounds__(256)
k_gemm128(const float* __restrict__ X, const unsigned short* __restrict__ Wf,
          const float* __restrict__ bias, float* __restrict__ Y, int M,
          const float* __restrict__ skipv) {
  __shared__ unsigned short sW[16384];               // 32 KB fragment image

  // cooperative global -> LDS stage of the whole weight matrix (reused by all
  // 8 waves x all k-steps); 256 threads x 8 x uint4
  {
    const uint4* gsrc = (const uint4*)Wf;            // 2048 x uint4
    uint4* ldst = (uint4*)sW;
#pragma unroll
    for (int i = 0; i < 8; ++i)
      ldst[threadIdx.x + 256 * i] = gsrc[threadIdx.x + 256 * i];
  }
  __syncthreads();

  const int wave = threadIdx.x >> 5;
  const int lane = threadIdx.x & 31;
  const int rowbase = blockIdx.x * 128 + wave * 16;
  int arow = rowbase + (lane & 15);
  if (arow >= M) arow = M - 1;                       // clamp; stores are guarded
  if (arow < 0) arow = 0;
  const float4* xp = (const float4*)(X + (size_t)arow * CC);
  const int b0 = (lane < 16) ? 0 : 2;                // ISA A-frag K interleave (float4 units)

  // ---- load the whole per-lane A row (16 x float4) in one clause ----------
  float4 raw[16];
#pragma unroll
  for (int s = 0; s < 4; ++s) {
    raw[s * 4 + 0] = xp[s * 8 + b0 + 0];
    raw[s * 4 + 1] = xp[s * 8 + b0 + 1];
    raw[s * 4 + 2] = xp[s * 8 + b0 + 4];
    raw[s * 4 + 3] = xp[s * 8 + b0 + 5];
  }

  // ---- convert all four A fragments (v_cvt_pk_bf16_f32) -------------------
  FragBF afrag[4];
#pragma unroll
  for (int s = 0; s < 4; ++s) {
    const float* fp = (const float*)&raw[s * 4];
#pragma unroll
    for (int i = 0; i < 16; ++i) {
      float v = fp[i];
      if (PRE_GELU) v = gelu_tanh(v);
      afrag[s].v[i] = (bf16_t)v;
    }
  }

  v8f acc[8];
#pragma unroll
  for (int j = 0; j < 8; ++j)
#pragma unroll
    for (int i = 0; i < 8; ++i) acc[j][i] = 0.0f;

  const uint4* lw = (const uint4*)sW;

#pragma unroll
  for (int s = 0; s < 4; ++s) {                      // K = 4 x 32
#pragma unroll
    for (int j = 0; j < 8; ++j) {                    // N = 8 x 16
      FragBF b;
      const int fidx = ((s * 8 + j) * 32 + lane) * 2;  // uint4 units
      b.q[0] = lw[fidx];                             // ds_load_b128 x2
      b.q[1] = lw[fidx + 1];
      acc[j] = __builtin_amdgcn_wmma_f32_16x16x32_bf16(
          false, afrag[s].v, false, b.v, (short)0, acc[j], false, false);
    }
  }

  float g = 0.f, omg = 0.f;
  if (SKIP_MIX) { float sv = *skipv; g = 1.0f / (1.0f + __expf(-sv)); omg = 1.0f - g; }
  const int col16 = lane & 15;
  const int rowadd = (lane < 16) ? 0 : 8;            // ISA C/D layout
#pragma unroll
  for (int j = 0; j < 8; ++j) {
    const int col = j * 16 + col16;
    const float bj = bias[col];
#pragma unroll
    for (int i = 0; i < 8; ++i) {
      const int r = rowbase + rowadd + i;
      if (r < M) {
        float v = acc[j][i] + bj;
        if (POST_RELU) v = fmaxf(v, 0.0f);
        if (SKIP_MIX)  v = g * v + omg * Y[(size_t)r * CC + col];
        Y[(size_t)r * CC + col] = v;
      }
    }
  }
}

// ---- effective weights: W_eff[r][h*32+f] = sum_d W[r][h*32+d] * rel[h][d][f]
__global__ void k_eff_w(const float* __restrict__ kW, const float* __restrict__ vW,
                        const float* __restrict__ relA, const float* __restrict__ relM,
                        float* __restrict__ effW) {
  int idx = blockIdx.x * blockDim.x + threadIdx.x;
  if (idx >= 12 * 16384) return;
  int m = idx >> 14;
  int rn = idx & 16383;
  int r = rn >> 7, n = rn & 127;
  int h = n >> 5, f = n & 31;
  int kv = m / 6, le = m % 6, l = le / 3, e = le % 3;
  int st = (e == 0) ? 1 : 0;
  const float* W = (kv ? vW : kW) + ((size_t)(l * TT + st) << 14);
  const float* R = (kv ? relM : relA) + ((size_t)((l * NEDGE + e) * HH + h)) * (DD * DD);
  float s = 0.f;
  for (int d = 0; d < DD; ++d) s += W[r * CC + h * DD + d] * R[d * DD + f];
  effW[idx] = s;
}

__global__ void k_eff_b(const float* __restrict__ kb, const float* __restrict__ vb,
                        const float* __restrict__ relA, const float* __restrict__ relM,
                        float* __restrict__ effB) {
  int idx = blockIdx.x * blockDim.x + threadIdx.x;
  if (idx >= 12 * 128) return;
  int m = idx >> 7, n = idx & 127;
  int h = n >> 5, f = n & 31;
  int kv = m / 6, le = m % 6, l = le / 3, e = le % 3;
  int st = (e == 0) ? 1 : 0;
  const float* bsrc = (kv ? vb : kb) + (size_t)(l * TT + st) * CC;
  const float* R = (kv ? relM : relA) + ((size_t)((l * NEDGE + e) * HH + h)) * (DD * DD);
  float s = 0.f;
  for (int d = 0; d < DD; ++d) s += bsrc[h * DD + d] * R[d * DD + f];
  effB[idx] = s;
}

// ---- pack 128x128 fp32 row-major weight into bf16 WMMA B-fragment order
__global__ void k_pack(const float* __restrict__ W, unsigned short* __restrict__ F) {
  int idx = blockIdx.x * blockDim.x + threadIdx.x;
  if (idx >= 16384) return;
  int k = idx >> 7, n = idx & 127;
  int s = k >> 5, kk = k & 31;
  int j = n >> 4;
  int lane = (n & 15) + (((kk >> 3) & 1) << 4);  // lanes 0-15: K 0-7,16-23; 16-31: K 8-15,24-31
  int elem = (kk & 7) + ((kk >> 4) << 3);
  F[(((s * 8 + j) * 32 + lane) << 4) + elem] = f2bf_bits(W[idx]);
}

__global__ void k_fill(float* __restrict__ p, long long n, float v) {
  long long i = (long long)blockIdx.x * blockDim.x + threadIdx.x;
  if (i < n) p[i] = v;
}

// ---- attention pass A: alpha = <q[dst,h], ke[src,h]> * relP[h] / sqrt(D); seg-max
__global__ void k_alpha(const int* __restrict__ ei, int E,
                        const float* __restrict__ q, const float* __restrict__ ke,
                        const float* __restrict__ relP,
                        float* __restrict__ ab, float* __restrict__ mbuf) {
  int idx = blockIdx.x * blockDim.x + threadIdx.x;
  if (idx >= E * HH) return;
  int e = idx >> 2, h = idx & 3;
  int src = ei[e], dst = ei[E + e];
  const float4* qp = (const float4*)(q + (size_t)dst * CC + h * DD);
  const float4* kp = (const float4*)(ke + (size_t)src * CC + h * DD);
  float dot = 0.f;
#pragma unroll
  for (int i = 0; i < 8; ++i) {
    float4 a = qp[i], b = kp[i];
    dot += a.x * b.x + a.y * b.y + a.z * b.z + a.w * b.w;
  }
  float al = dot * relP[h] * 0.17677669529663687f;  // 1/sqrt(32)
  ab[idx] = al;
  atomicMaxF(mbuf + (size_t)dst * HH + h, al);
}

// ---- pass B: exp(alpha - max); seg-sum
__global__ void k_expsum(const int* __restrict__ ei, int E,
                         float* __restrict__ ab, const float* __restrict__ mbuf,
                         float* __restrict__ denom) {
  int idx = blockIdx.x * blockDim.x + threadIdx.x;
  if (idx >= E * HH) return;
  int e = idx >> 2, h = idx & 3;
  int dst = ei[E + e];
  float ex = __expf(ab[idx] - mbuf[(size_t)dst * HH + h]);
  ab[idx] = ex;
  atomicAdd(denom + (size_t)dst * HH + h, ex);
}

// ---- pass C: agg[dst] += att * ve[src]
__global__ void k_agg(const int* __restrict__ ei, int E,
                      const float* __restrict__ ab, const float* __restrict__ denom,
                      const float* __restrict__ ve, float* __restrict__ agg) {
  int idx = blockIdx.x * blockDim.x + threadIdx.x;
  if (idx >= E * HH) return;
  int e = idx >> 2, h = idx & 3;
  int src = ei[e], dst = ei[E + e];
  float att = ab[idx] / (denom[(size_t)dst * HH + h] + 1e-16f);
  const float4* vp = (const float4*)(ve + (size_t)src * CC + h * DD);
  float* ap = agg + (size_t)dst * CC + h * DD;
#pragma unroll
  for (int i = 0; i < 8; ++i) {
    float4 v = vp[i];
    atomicAdd(ap + 4 * i + 0, att * v.x);
    atomicAdd(ap + 4 * i + 1, att * v.y);
    atomicAdd(ap + 4 * i + 2, att * v.z);
    atomicAdd(ap + 4 * i + 3, att * v.w);
  }
}

__global__ void k_gather(const float* __restrict__ x0, const int* __restrict__ slices,
                         float* __restrict__ G, int B) {
  int idx = blockIdx.x * blockDim.x + threadIdx.x;
  if (idx >= B * CC) return;
  int b = idx >> 7, c = idx & 127;
  int r = slices[b + 1] - 1;
  G[idx] = x0[(size_t)r * CC + c];
}

// ---------------------------------------------------------------------------
extern "C" void kernel_launch(void* const* d_in, const int* in_sizes, int n_in,
                              void* d_out, int out_size, void* d_ws, size_t ws_size,
                              hipStream_t stream) {
  const float* x_visit = (const float*)d_in[0];
  const float* x_code  = (const float*)d_in[1];
  const int* edge_in   = (const int*)d_in[2];
  const int* edge_has  = (const int*)d_in[3];
  const int* edge_next = (const int*)d_in[4];
  const int* slices    = (const int*)d_in[5];
  const float* lin0_W  = (const float*)d_in[6];
  const float* lin0_b  = (const float*)d_in[7];
  const float* kW  = (const float*)d_in[8];
  const float* kb  = (const float*)d_in[9];
  const float* qW  = (const float*)d_in[10];
  const float* qb  = (const float*)d_in[11];
  const float* vW  = (const float*)d_in[12];
  const float* vb  = (const float*)d_in[13];
  const float* aW  = (const float*)d_in[14];
  const float* ab_ = (const float*)d_in[15];
  const float* skip = (const float*)d_in[16];
  const float* relA = (const float*)d_in[17];
  const float* relM = (const float*)d_in[18];
  const float* relP = (const float*)d_in[19];
  const float* linW = (const float*)d_in[20];
  const float* linb = (const float*)d_in[21];

  const int NV = in_sizes[0] / CC;
  const int NC = in_sizes[1] / CC;
  const int E_IN   = in_sizes[2] / 2;
  const int E_HAS  = in_sizes[3] / 2;
  const int E_NEXT = in_sizes[4] / 2;
  const int B = in_sizes[5] - 1;
  const int Nmax = (NV > NC) ? NV : NC;
  int Emax = E_IN; if (E_HAS > Emax) Emax = E_HAS; if (E_NEXT > Emax) Emax = E_NEXT;

  // workspace carve-out
  char* w = (char*)d_ws;
  size_t off = 0;
  auto carve = [&](size_t bytes) -> void* {
    void* p = (void*)(w + off);
    off = (off + bytes + 255) & ~(size_t)255;
    return p;
  };
  float* effW  = (float*)carve((size_t)12 * 16384 * 4);
  float* effB  = (float*)carve((size_t)12 * 128 * 4);
  unsigned short* frag = (unsigned short*)carve((size_t)23 * 16384 * 2);
  float* x0   = (float*)carve((size_t)NV * CC * 4);
  float* x1   = (float*)carve((size_t)NC * CC * 4);
  float* q0   = (float*)carve((size_t)NV * CC * 4);
  float* q1   = (float*)carve((size_t)NC * CC * 4);
  float* keb  = (float*)carve((size_t)Nmax * CC * 4);
  float* veb  = (float*)carve((size_t)Nmax * CC * 4);
  float* agg0 = (float*)carve((size_t)NV * CC * 4);
  float* agg1 = (float*)carve((size_t)NC * CC * 4);
  float* mbuf = (float*)carve((size_t)Nmax * HH * 4);
  float* dbuf = (float*)carve((size_t)Nmax * HH * 4);
  float* abuf = (float*)carve((size_t)Emax * HH * 4);
  float* G    = (float*)carve((size_t)B * CC * 4);
  (void)ws_size; (void)n_in; (void)out_size;

  auto fragp = [&](int i) { return frag + (size_t)i * 16384; };

  // ---- prep: effective k/v weights+biases, pack all 23 weight matrices ----
  k_eff_w<<<(12 * 16384 + 255) / 256, 256, 0, stream>>>(kW, vW, relA, relM, effW);
  k_eff_b<<<(12 * 128 + 255) / 256, 256, 0, stream>>>(kb, vb, relA, relM, effB);

  // frag index map: 0-1 lin0[t]; 2-5 qW[l,t]; 6-9 aW[l,t];
  //                 10-15 kW_eff[l,e]; 16-21 vW_eff[l,e]; 22 linW
  for (int t = 0; t < TT; ++t)
    k_pack<<<64, 256, 0, stream>>>(lin0_W + (size_t)t * 16384, fragp(t));
  for (int i = 0; i < LL * TT; ++i) {
    k_pack<<<64, 256, 0, stream>>>(qW + (size_t)i * 16384, fragp(2 + i));
    k_pack<<<64, 256, 0, stream>>>(aW + (size_t)i * 16384, fragp(6 + i));
  }
  for (int i = 0; i < LL * NEDGE; ++i) {
    k_pack<<<64, 256, 0, stream>>>(effW + (size_t)i * 16384, fragp(10 + i));
    k_pack<<<64, 256, 0, stream>>>(effW + (size_t)(6 + i) * 16384, fragp(16 + i));
  }
  k_pack<<<64, 256, 0, stream>>>(linW, fragp(22));

  auto gb = [](int M) { return dim3((unsigned)((M + 127) / 128)); };

  // ---- input projection + relu ----
  k_gemm128<0, 1, 0><<<gb(NV), 256, 0, stream>>>(x_visit, fragp(0), lin0_b, x0, NV, nullptr);
  k_gemm128<0, 1, 0><<<gb(NC), 256, 0, stream>>>(x_code, fragp(1), lin0_b + CC, x1, NC, nullptr);

  float* xs[2]  = {x0, x1};
  float* qs[2]  = {q0, q1};
  float* aggs[2] = {agg0, agg1};
  const int Ns[2] = {NV, NC};

  struct Rel { const int* ei; int E; int st; int dt; };
  Rel rels[NEDGE] = {{edge_in, E_IN, 1, 0}, {edge_has, E_HAS, 0, 1}, {edge_next, E_NEXT, 0, 0}};

  for (int l = 0; l < LL; ++l) {
    // q projections for both node types
    for (int t = 0; t < TT; ++t)
      k_gemm128<0, 0, 0><<<gb(Ns[t]), 256, 0, stream>>>(
          xs[t], fragp(2 + l * TT + t), qb + (size_t)(l * TT + t) * CC, qs[t], Ns[t], nullptr);
    // zero aggregation buffers
    for (int t = 0; t < TT; ++t) {
      long long n = (long long)Ns[t] * CC;
      k_fill<<<(unsigned)((n + 255) / 256), 256, 0, stream>>>(aggs[t], n, 0.0f);
    }
    for (int e = 0; e < NEDGE; ++e) {
      const Rel& R = rels[e];
      const int Ms = Ns[R.st], Md = Ns[R.dt];
      // relation-transformed keys/values via folded weights (WMMA GEMMs)
      k_gemm128<0, 0, 0><<<gb(Ms), 256, 0, stream>>>(
          xs[R.st], fragp(10 + l * NEDGE + e), effB + (size_t)(l * NEDGE + e) * CC,
          keb, Ms, nullptr);
      k_gemm128<0, 0, 0><<<gb(Ms), 256, 0, stream>>>(
          xs[R.st], fragp(16 + l * NEDGE + e), effB + (size_t)(6 + l * NEDGE + e) * CC,
          veb, Ms, nullptr);
      // segment softmax (per relation, as in reference)
      long long nm = (long long)Md * HH;
      k_fill<<<(unsigned)((nm + 255) / 256), 256, 0, stream>>>(mbuf, nm, -INFINITY);
      k_fill<<<(unsigned)((nm + 255) / 256), 256, 0, stream>>>(dbuf, nm, 0.0f);
      const unsigned eg = (unsigned)((R.E * HH + 255) / 256);
      k_alpha<<<eg, 256, 0, stream>>>(R.ei, R.E, qs[R.dt], keb,
                                      relP + (size_t)(l * NEDGE + e) * HH, abuf, mbuf);
      k_expsum<<<eg, 256, 0, stream>>>(R.ei, R.E, abuf, mbuf, dbuf);
      k_agg<<<eg, 256, 0, stream>>>(R.ei, R.E, abuf, dbuf, veb, aggs[R.dt]);
    }
    // output projection: x = sigmoid(skip)*(gelu(agg)@aW+ab) + (1-s)*x, in place
    for (int t = 0; t < TT; ++t)
      k_gemm128<1, 0, 1><<<gb(Ns[t]), 256, 0, stream>>>(
          aggs[t], fragp(6 + l * TT + t), ab_ + (size_t)(l * TT + t) * CC,
          xs[t], Ns[t], skip + (l * TT + t));
  }

  // ---- last visit per patient, final linear ----
  k_gather<<<(unsigned)((B * CC + 255) / 256), 256, 0, stream>>>(x0, slices, G, B);
  k_gemm128<0, 0, 0><<<gb(B), 256, 0, stream>>>(G, fragp(22), linb, (float*)d_out, B, nullptr);
}
